// GlobalBatchTopKMatryoshkaSAE_9156870275254
// MI455X (gfx1250) — matrix operand order repaired and checked by reading, not compiled, not run on hardware
//
#include <hip/hip_runtime.h>
#include <hip/hip_bf16.h>
#include <math.h>

// ---------------- static problem config ----------------
#define ACT   768
#define BATCH 2048
#define DICT  32768
#define KTOT  (32 * 2048)   // global top-k count
#define NG    5

// scalar workspace slots (float array; some slots hold uint bits)
#define S_SUMXN   0
#define S_SUMXN2  1
#define S_L1      2
#define S_CNTPOS  3
#define S_MINPOS  4   // uint bits, atomicMin
#define S_TCUT    5
#define S_L2      6   // 6..11 : six l2 sums
#define S_AUXSQ   12
#define S_AUXABS  13
#define S_NDEAD   14
#define NSCAL     16

typedef __attribute__((ext_vector_type(16))) __bf16          v16bf;
typedef __attribute__((ext_vector_type(8)))  float           v8f;
typedef __attribute__((ext_vector_type(16))) unsigned short  v16us;
typedef __attribute__((ext_vector_type(8)))  unsigned short  v8us;

// async global->LDS copy path (guarded: falls back to vector load/store staging)
#if defined(__gfx1250__) && __has_builtin(__builtin_amdgcn_global_load_async_to_lds_b128) && __has_builtin(__builtin_amdgcn_s_wait_asynccnt)
#define USE_ASYNC_LDS 1

// builtin param type (from hipcc diagnostic): 'int __attribute__((vector_size(16))) __device__ *'
typedef int b128_t __attribute__((vector_size(16)));
typedef __attribute__((address_space(1))) b128_t gb128_t;
typedef __attribute__((address_space(3))) b128_t lb128_t;

// generic->AS1: identical representation. generic LDS ptr low 32 bits == AS3 offset
// (flat->LDS truncation rule), so integer round-trip casts are semantically exact.
__device__ __forceinline__ void async_copy_b128(const void* g, void* l) {
  __builtin_amdgcn_global_load_async_to_lds_b128(
      (gb128_t*)(unsigned long long)(size_t)g,
      (lb128_t*)(unsigned)(size_t)l, 0, 0);
}
#endif

// float -> bf16 (round to nearest even), as raw u16
__device__ __forceinline__ unsigned short f2bf(float f) {
  unsigned u = __builtin_bit_cast(unsigned, f);
  unsigned r = (u + 0x7FFFu + ((u >> 16) & 1u)) >> 16;
  return (unsigned short)r;
}

__device__ __forceinline__ float blockReduceSum(float v, float* sbuf) {
  int tid = threadIdx.x;
  sbuf[tid] = v;
  __syncthreads();
  for (int s = blockDim.x >> 1; s > 0; s >>= 1) {
    if (tid < s) sbuf[tid] += sbuf[tid + s];
    __syncthreads();
  }
  float r = sbuf[0];
  __syncthreads();
  return r;
}

// ---------------- small utility kernels ----------------
__global__ void k_zero(float* p, long n) {
  long i = (long)blockIdx.x * blockDim.x + threadIdx.x;
  long stride = (long)gridDim.x * blockDim.x;
  for (; i < n; i += stride) p[i] = 0.0f;
}

__global__ void k_init_scal(float* scal) {
  int t = threadIdx.x;
  if (t < NSCAL) scal[t] = 0.0f;
  if (t == 0) ((unsigned*)scal)[S_MINPOS] = 0x7F800000u;  // +inf bits
}

// ---------------- f32 -> bf16 conversion + WMMA B-tile swizzle ----------------
// dst layout: 32(K)x16(N) tiles, each stored as [lane][e] (1KB contiguous):
//   lane = (n%16) + 16*((k%32)/16), e = k%16
//   flatTile = (n/16)*(K/32) + (k/32);  off = (flatTile*32 + lane)*16 + e
__global__ __launch_bounds__(256)
void k_cvt_swz(const float* __restrict__ src, unsigned short* __restrict__ dst,
               int K, int N) {
  long total = (long)K * N;
  long idx = (long)blockIdx.x * blockDim.x + threadIdx.x;
  long stride = (long)gridDim.x * blockDim.x;
  int ktiles = K >> 5;
  for (; idx < total; idx += stride) {
    int n = (int)(idx % N);
    int k = (int)(idx / N);
    int tk = k >> 5, kk = k & 31;
    int tn = n >> 4, nn = n & 15;
    int lane = nn + ((kk >> 4) << 4);
    int e = kk & 15;
    long off = ((((long)tn * ktiles + tk) << 5) + lane) * 16 + e;
    dst[off] = f2bf(src[idx]);
  }
}

// ---------------- preprocess: per-row normalize ----------------
__global__ __launch_bounds__(256)
void k_preprocess(const float* __restrict__ x, const float* __restrict__ b_dec,
                  float* __restrict__ xn, unsigned short* __restrict__ xcb,
                  float* __restrict__ rmean, float* __restrict__ rstd,
                  float* __restrict__ scal) {
  __shared__ float sbuf[256];
  int row = blockIdx.x;
  int tid = threadIdx.x;
  const float* xr = x + (long)row * ACT;
  float s = 0.f, s2 = 0.f;
  for (int c = tid; c < ACT; c += blockDim.x) {
    float v = xr[c];
    s += v;
    s2 += v * v;
  }
  float S  = blockReduceSum(s, sbuf);
  float S2 = blockReduceSum(s2, sbuf);
  float mean = S / (float)ACT;
  float var  = (S2 - S * S / (float)ACT) / (float)(ACT - 1);  // unbiased
  float sd   = sqrtf(fmaxf(var, 0.0f));
  if (tid == 0) { rmean[row] = mean; rstd[row] = sd; }
  float inv = 1.0f / (sd + 1e-5f);
  float ls = 0.f, ls2 = 0.f;
  for (int c = tid; c < ACT; c += blockDim.x) {
    float v  = (xr[c] - mean) * inv;
    xn[(long)row * ACT + c] = v;
    xcb[(long)row * ACT + c] = f2bf(v - b_dec[c]);  // encoder input, bf16
    ls += v;
    ls2 += v * v;
  }
  float GS  = blockReduceSum(ls, sbuf);
  float GS2 = blockReduceSum(ls2, sbuf);
  if (tid == 0) {
    atomicAdd(&scal[S_SUMXN],  GS);
    atomicAdd(&scal[S_SUMXN2], GS2);
  }
}

// ---------------- bf16 WMMA GEMM: C[M,N] = (relu)(A[M,K] * Bsw[K,N]) ----------------
// A: row-major bf16. Bsw: B-fragment tiled bf16 (see k_cvt_swz).
// Block tile 128x64, 8 waves, each wave 32x32 (2x2 wmma tiles), K step 32.
// Double-buffered LDS; async global->LDS copies pipelined against WMMA.
#define TM 128
#define TN 64
#define TK 32

__global__ __launch_bounds__(256)
void k_gemm_bf16(const unsigned short* __restrict__ A, int lda,
                 const unsigned short* __restrict__ Bsw, int bTileStride, int tkOff,
                 float* __restrict__ C, int ldc,
                 int M, int N, int K, int doRelu) {
  __shared__ unsigned short As[2][TM * TK];   // [r*TK + k], 64B rows (8KB each)
  __shared__ unsigned short BsF[2][4 * 512];  // 4 tiles x (32 lanes x 16 e) (4KB each)

  const int row0 = blockIdx.y * TM;
  const int col0 = blockIdx.x * TN;
  const int tn0  = col0 >> 4;
  const int tid  = threadIdx.x;
  const int wave = tid >> 5;
  const int lane = tid & 31;
  const int wm = wave & 3;   // 0..3 -> 32-row strip
  const int wn = wave >> 2;  // 0..1 -> 32-col strip
  const int l15    = lane & 15;
  const int hiHalf = lane >> 4;  // 0 or 1

  v8f acc[2][2];
  #pragma unroll
  for (int i = 0; i < 2; i++)
    #pragma unroll
    for (int j = 0; j < 2; j++)
      #pragma unroll
      for (int r = 0; r < 8; r++) acc[i][j][r] = 0.0f;

  // A tile: 128 rows x 64B = 512 16B chunks -> 2 per thread
  const int ar0 = (2 * tid) >> 2,     aseg0 = (2 * tid) & 3;
  const int ar1 = (2 * tid + 1) >> 2, aseg1 = (2 * tid + 1) & 3;
  // B tiles: 4 x 1KB = 256 16B chunks -> 1 per thread
  const int btile = tid >> 6, bchunk = tid & 63;

  auto stage = [&](int k0, int buf) {
    const unsigned short* gA0 = A + (long)(row0 + ar0) * lda + k0 + aseg0 * 8;
    const unsigned short* gA1 = A + (long)(row0 + ar1) * lda + k0 + aseg1 * 8;
    long tileBase = (((long)(tn0 + btile) * bTileStride + tkOff + (k0 >> 5)) << 9);
    const unsigned short* gB = Bsw + tileBase + bchunk * 8;
    unsigned short* lA0 = &As[buf][ar0 * TK + aseg0 * 8];
    unsigned short* lA1 = &As[buf][ar1 * TK + aseg1 * 8];
    unsigned short* lB  = &BsF[buf][tid * 8];
#ifdef USE_ASYNC_LDS
    async_copy_b128(gA0, lA0);
    async_copy_b128(gA1, lA1);
    async_copy_b128(gB, lB);
#else
    *(v8us*)lA0 = *(const v8us*)gA0;
    *(v8us*)lA1 = *(const v8us*)gA1;
    *(v8us*)lB  = *(const v8us*)gB;
#endif
  };

  stage(0, 0);
#ifdef USE_ASYNC_LDS
  __builtin_amdgcn_s_wait_asynccnt(0);
#endif
  __syncthreads();

  int cur = 0;
  for (int k0 = 0; k0 < K; k0 += TK) {
    int nxt = cur ^ 1;
    if (k0 + TK < K) stage(k0 + TK, nxt);  // pipeline next tile behind compute

    // A fragments: contiguous K order matching B swizzle, lane<16 holds K {0..15}
    // -> one contiguous 32B run per lane within a 64B LDS row.
    v16bf aF[2], bF[2];
    #pragma unroll
    for (int sm = 0; sm < 2; sm++) {
      int rbase = wm * 32 + sm * 16 + l15;
      const v8us* ap = (const v8us*)(&As[cur][rbase * TK + hiHalf * 16]);
      v8us lo = ap[0];   // K = hi*16 .. hi*16+7
      v8us hi = ap[1];   // K = hi*16+8 .. hi*16+15
      v16us au = __builtin_shufflevector(lo, hi, 0, 1, 2, 3, 4, 5, 6, 7,
                                         8, 9, 10, 11, 12, 13, 14, 15);
      aF[sm] = __builtin_bit_cast(v16bf, au);
    }
    // B fragments: pre-swizzled tiles -> one contiguous 32B run per lane.
    #pragma unroll
    for (int sn = 0; sn < 2; sn++) {
      int t = wn * 2 + sn;
      v16us bu = *(const v16us*)(&BsF[cur][t * 512 + lane * 16]);
      bF[sn] = __builtin_bit_cast(v16bf, bu);
    }

    #pragma unroll
    for (int sm = 0; sm < 2; sm++)
      #pragma unroll
      for (int sn = 0; sn < 2; sn++)
        acc[sm][sn] = __builtin_amdgcn_wmma_f32_16x16x32_bf16(
            false, aF[sm], false, bF[sn], (short)0, acc[sm][sn], false, false);

#ifdef USE_ASYNC_LDS
    __builtin_amdgcn_s_wait_asynccnt(0);
#endif
    __syncthreads();
    cur = nxt;
  }

  // store C per 16x16 f32 C/D layout
  #pragma unroll
  for (int sm = 0; sm < 2; sm++) {
    #pragma unroll
    for (int sn = 0; sn < 2; sn++) {
      #pragma unroll
      for (int r = 0; r < 8; r++) {
        int mrow = row0 + wm * 32 + sm * 16 + (hiHalf ? (8 + r) : r);
        int ncol = col0 + wn * 32 + sn * 16 + l15;
        float v = acc[sm][sn][r];
        if (doRelu) v = fmaxf(v, 0.0f);
        C[(long)mrow * ldc + ncol] = v;
      }
    }
  }
}

// ---------------- global top-k via 65536-bin float-bit histogram ----------------
__global__ void k_hist_build(const float* __restrict__ acts, unsigned* __restrict__ hist, long n) {
  long i = (long)blockIdx.x * blockDim.x + threadIdx.x;
  long stride = (long)gridDim.x * blockDim.x;
  for (; i < n; i += stride) {
    float v = acts[i];
    if (v > 0.0f) {
      unsigned b = __builtin_bit_cast(unsigned, v) >> 16;  // monotone for +floats
      atomicAdd(&hist[b], 1u);
    }
  }
}

__global__ void k_hist_scan(const unsigned* __restrict__ hist, float* __restrict__ scal) {
  if (blockIdx.x == 0 && threadIdx.x == 0) {
    unsigned acc = 0;
    unsigned cut_bits = 1u;  // tiny positive -> keep all positives if fewer than k
    for (int b = 65535; b >= 1; --b) {
      acc += hist[b];
      if (acc >= (unsigned)KTOT) { cut_bits = ((unsigned)b) << 16; break; }
    }
    scal[S_TCUT] = __builtin_bit_cast(float, cut_bits);
  }
}

__global__ __launch_bounds__(256)
void k_apply_topk(const float* __restrict__ acts, float* __restrict__ topk,
                  unsigned short* __restrict__ tkb, float* __restrict__ colSum,
                  float* __restrict__ scal) {
  __shared__ float sbuf[256];
  unsigned* us = (unsigned*)scal;
  float tcut = scal[S_TCUT];
  long n = (long)BATCH * DICT;
  long i = (long)blockIdx.x * blockDim.x + threadIdx.x;
  long stride = (long)gridDim.x * blockDim.x;
  float l1 = 0.f, cnt = 0.f;
  unsigned mn = 0x7F800000u;
  for (; i < n; i += stride) {
    float v = acts[i];
    bool keep = (v > 0.0f) && (v >= tcut);
    topk[i] = keep ? v : 0.0f;
    tkb[i] = keep ? f2bf(v) : (unsigned short)0;
    if (keep) {
      int col = (int)(i & (DICT - 1));
      atomicAdd(&colSum[col], v);
      l1 += v;
      cnt += 1.0f;
      unsigned bv = __builtin_bit_cast(unsigned, v);
      mn = bv < mn ? bv : mn;
    }
  }
  if (mn < 0x7F800000u) atomicMin(&us[S_MINPOS], mn);
  l1  = blockReduceSum(l1, sbuf);
  cnt = blockReduceSum(cnt, sbuf);
  if (threadIdx.x == 0) {
    if (l1 != 0.f)  atomicAdd(&scal[S_L1], l1);
    if (cnt != 0.f) atomicAdd(&scal[S_CNTPOS], cnt);
  }
}

__global__ void k_nb_dead(const float* __restrict__ colSum, const float* __restrict__ nba,
                          float* __restrict__ nb, float* __restrict__ scal) {
  int j = blockIdx.x * blockDim.x + threadIdx.x;
  if (j < DICT) {
    float v = (colSum[j] > 0.0f) ? 0.0f : (nba[j] + 1.0f);
    nb[j] = v;
    if (v > 5.0f) atomicAdd(&scal[S_NDEAD], 1.0f);
  }
}

// ---------------- cumulative matryoshka reconstruction + L2 terms ----------------
__global__ __launch_bounds__(256)
void k_recon(const float* __restrict__ xn, const float* __restrict__ P,
             const float* __restrict__ b_dec, const float* __restrict__ rmean,
             const float* __restrict__ rstd, float* __restrict__ residual,
             float* __restrict__ sae_out, float* __restrict__ scal) {
  __shared__ float sbuf[256];
  int row = blockIdx.x;
  int tid = threadIdx.x;
  float part[NG + 1];
  #pragma unroll
  for (int t = 0; t <= NG; t++) part[t] = 0.0f;
  float mean = rmean[row], sd = rstd[row];
  for (int c = tid; c < ACT; c += blockDim.x) {
    float xv = xn[(long)row * ACT + c];
    float accv = b_dec[c];
    float d0 = accv - xv;
    part[0] += d0 * d0;
    #pragma unroll
    for (int g = 0; g < NG; g++) {
      accv += P[((long)g * BATCH + row) * ACT + c];
      float d = accv - xv;
      part[g + 1] += d * d;
    }
    residual[(long)row * ACT + c] = xv - accv;
    sae_out[(long)row * ACT + c] = accv * sd + mean;
  }
  #pragma unroll
  for (int t = 0; t <= NG; t++) {
    float s = blockReduceSum(part[t], sbuf);
    if (tid == 0) atomicAdd(&scal[S_L2 + t], s);
  }
}

// ---------------- per-row aux top-512 among dead features -> bf16 ----------------
__global__ __launch_bounds__(256)
void k_aux_select(const float* __restrict__ acts, const float* __restrict__ nb,
                  unsigned short* __restrict__ axb) {
  __shared__ unsigned hist[2048];
  __shared__ float cutS;
  int row = blockIdx.x;
  int tid = threadIdx.x;
  for (int b = tid; b < 2048; b += blockDim.x) hist[b] = 0u;
  __syncthreads();
  const float* arow = acts + (long)row * DICT;
  for (int j = tid; j < DICT; j += blockDim.x) {
    float v = arow[j];
    if (nb[j] >= 5.0f && v > 0.0f) {
      atomicAdd(&hist[__builtin_bit_cast(unsigned, v) >> 20], 1u);
    }
  }
  __syncthreads();
  if (tid == 0) {
    unsigned acc = 0;
    unsigned cb = 1u;
    for (int b = 2047; b >= 1; --b) {
      acc += hist[b];
      if (acc >= 512u) { cb = ((unsigned)b) << 20; break; }
    }
    cutS = __builtin_bit_cast(float, cb);
  }
  __syncthreads();
  float cut = cutS;
  unsigned short* orow = axb + (long)row * DICT;
  for (int j = tid; j < DICT; j += blockDim.x) {
    float v = arow[j];
    bool keep = (nb[j] >= 5.0f) && (v > 0.0f) && (v >= cut);
    orow[j] = keep ? f2bf(v) : (unsigned short)0;
  }
}

__global__ __launch_bounds__(256)
void k_aux_mse(const float* __restrict__ auxR, const float* __restrict__ residual,
               float* __restrict__ scal) {
  __shared__ float sbuf[256];
  long n = (long)BATCH * ACT;
  long i = (long)blockIdx.x * blockDim.x + threadIdx.x;
  long stride = (long)gridDim.x * blockDim.x;
  float sq = 0.f, ab = 0.f;
  for (; i < n; i += stride) {
    float a = auxR[i];
    float d = a - residual[i];
    sq += d * d;
    ab += fabsf(a);
  }
  sq = blockReduceSum(sq, sbuf);
  ab = blockReduceSum(ab, sbuf);
  if (threadIdx.x == 0) {
    atomicAdd(&scal[S_AUXSQ], sq);
    atomicAdd(&scal[S_AUXABS], ab);
  }
}

__global__ void k_finalize(const float* __restrict__ scal,
                           const float* __restrict__ threshold_in,
                           float* __restrict__ outScal) {
  if (blockIdx.x == 0 && threadIdx.x == 0) {
    const float NEL = (float)BATCH * (float)ACT;
    float terms[NG + 1];
    float total = 0.f;
    for (int t = 0; t <= NG; t++) { terms[t] = scal[S_L2 + t] / NEL; total += terms[t]; }
    float mean_l2 = total / (float)(NG + 1);
    float l2min = terms[1], l2max = terms[1];
    for (int t = 2; t <= NG; t++) {
      l2min = fminf(l2min, terms[t]);
      l2max = fmaxf(l2max, terms[t]);
    }
    float l1n = scal[S_L1] / (float)BATCH;
    float l0n = scal[S_CNTPOS] / (float)BATCH;
    float l1_loss = 0.0003f * l1n;
    float aux_mse = scal[S_AUXSQ] / NEL;
    float aux_loss = (scal[S_AUXABS] > 0.0f) ? 0.03125f * aux_mse : 0.0f;
    float loss = mean_l2 + l1_loss + aux_loss;
    float S = scal[S_SUMXN], S2 = scal[S_SUMXN2];
    float xvar = (S2 - S * S / NEL) / (NEL - 1.0f);
    float fvu = mean_l2 / (xvar + 1e-10f);
    unsigned mb = ((const unsigned*)scal)[S_MINPOS];
    float minpos = __builtin_bit_cast(float, mb);
    float thr = threshold_in[0];
    float newthr = (scal[S_CNTPOS] > 0.0f) ? (0.99f * thr + 0.01f * minpos) : thr;
    outScal[0]  = loss;
    outScal[1]  = l1_loss;
    outScal[2]  = mean_l2;
    outScal[3]  = l2min;
    outScal[4]  = l2max;
    outScal[5]  = l0n;
    outScal[6]  = l1n;
    outScal[7]  = aux_loss;
    outScal[8]  = newthr;
    outScal[9]  = fvu;
    outScal[10] = scal[S_NDEAD];
  }
}

// ---------------- host-side orchestration ----------------
extern "C" void kernel_launch(void* const* d_in, const int* in_sizes, int n_in,
                              void* d_out, int out_size, void* d_ws, size_t ws_size,
                              hipStream_t stream) {
  const float* x      = (const float*)d_in[0];
  const float* W_enc  = (const float*)d_in[1];
  const float* W_dec  = (const float*)d_in[2];
  const float* b_dec  = (const float*)d_in[3];
  const float* nba    = (const float*)d_in[4];
  const float* thr_in = (const float*)d_in[5];

  float* out = (float*)d_out;
  float* sae_out   = out;                              // [2048*768]
  float* acts_topk = out + (long)BATCH * ACT;          // [2048*32768]
  float* out_scal  = acts_topk + (long)BATCH * DICT;   // 11 scalars

  // workspace layout (256B-aligned chunks)
  char* wsb = (char*)d_ws;
  auto alloc = [&](size_t bytes) -> char* {
    char* p = wsb;
    wsb += (bytes + 255) & ~(size_t)255;
    return p;
  };
  float*          XN    = (float*)alloc((size_t)BATCH * ACT * 4);
  float*          RMEAN = (float*)alloc((size_t)BATCH * 4);
  float*          RSTD  = (float*)alloc((size_t)BATCH * 4);
  float*          SCAL  = (float*)alloc(NSCAL * 4);
  float*          NB    = (float*)alloc((size_t)DICT * 4);
  float*          CSUM  = (float*)alloc((size_t)DICT * 4);
  unsigned*       HIST  = (unsigned*)alloc((size_t)65536 * 4);
  float*          ACTS  = (float*)alloc((size_t)BATCH * DICT * 4);
  float*          P     = (float*)alloc((size_t)NG * BATCH * ACT * 4);
  float*          RESID = (float*)alloc((size_t)BATCH * ACT * 4);
  float*          AUXR  = (float*)alloc((size_t)BATCH * ACT * 4);
  unsigned short* XCb   = (unsigned short*)alloc((size_t)BATCH * ACT * 2);
  unsigned short* WEb   = (unsigned short*)alloc((size_t)ACT * DICT * 2);   // swizzled
  unsigned short* WDb   = (unsigned short*)alloc((size_t)DICT * ACT * 2);   // swizzled
  unsigned short* TKb   = (unsigned short*)alloc((size_t)BATCH * DICT * 2);
  unsigned short* AXb   = (unsigned short*)alloc((size_t)BATCH * DICT * 2);

  static const int GRP_OFF[NG + 1] = {0, 2048, 4096, 8192, 16384, 32768};

  // 1) init reduction state + weight conversion (f32 -> bf16, B-fragment tiling)
  k_init_scal<<<1, 32, 0, stream>>>(SCAL);
  k_zero<<<256, 256, 0, stream>>>(CSUM, DICT);
  k_zero<<<256, 256, 0, stream>>>((float*)HIST, 65536);
  k_cvt_swz<<<2048, 256, 0, stream>>>(W_enc, WEb, ACT, DICT);
  k_cvt_swz<<<2048, 256, 0, stream>>>(W_dec, WDb, DICT, ACT);

  // 2) preprocess (xn f32 + encoder input bf16)
  k_preprocess<<<BATCH, 256, 0, stream>>>(x, b_dec, XN, XCb, RMEAN, RSTD, SCAL);

  // 3) encoder GEMM: ACTS = relu(XCb @ W_enc)   [2048 x 32768]
  k_gemm_bf16<<<dim3(DICT / TN, BATCH / TM), 256, 0, stream>>>(
      XCb, ACT, WEb, ACT / 32, 0, ACTS, DICT, BATCH, DICT, ACT, 1);

  // 4) global batch top-k
  k_hist_build<<<4096, 256, 0, stream>>>(ACTS, HIST, (long)BATCH * DICT);
  k_hist_scan<<<1, 32, 0, stream>>>(HIST, SCAL);
  k_apply_topk<<<4096, 256, 0, stream>>>(ACTS, acts_topk, TKb, CSUM, SCAL);

  // 5) dead-feature bookkeeping
  k_nb_dead<<<DICT / 256, 256, 0, stream>>>(CSUM, nba, NB, SCAL);

  // 6) matryoshka group decode GEMMs: P[g] = topk[:, s:e] @ W_dec[s:e, :]
  for (int g = 0; g < NG; g++) {
    int s = GRP_OFF[g];
    int gk = GRP_OFF[g + 1] - s;
    k_gemm_bf16<<<dim3(ACT / TN, BATCH / TM), 256, 0, stream>>>(
        TKb + s, DICT, WDb, DICT / 32, s / 32,
        P + (long)g * BATCH * ACT, ACT, BATCH, ACT, gk, 0);
  }

  // 7) cumulative reconstruction, residual, sae_out, L2 terms
  k_recon<<<BATCH, 256, 0, stream>>>(XN, P, b_dec, RMEAN, RSTD, RESID, sae_out, SCAL);

  // 8) aux: per-row top-512 among dead -> bf16, dense aux GEMM, aux mse
  k_aux_select<<<BATCH, 256, 0, stream>>>(ACTS, NB, AXb);
  k_gemm_bf16<<<dim3(ACT / TN, BATCH / TM), 256, 0, stream>>>(
      AXb, DICT, WDb, DICT / 32, 0, AUXR, ACT, BATCH, ACT, DICT, 0);
  k_aux_mse<<<1024, 256, 0, stream>>>(AUXR, RESID, SCAL);

  // 9) scalars
  k_finalize<<<1, 32, 0, stream>>>(SCAL, thr_in, out_scal);
}
